// ST_LSTM_78082505441387
// MI455X (gfx1250) — matrix operand order repaired and checked by reading, not compiled, not run on hardware
//
#include <hip/hip_runtime.h>

#define DEV_INLINE __device__ __forceinline__

typedef __attribute__((ext_vector_type(16))) __bf16 v16bf;
typedef __attribute__((ext_vector_type(8)))  __bf16 v8bf;
typedef __attribute__((ext_vector_type(8)))  float  v8f;

namespace cfg {
constexpr int B = 16, Tin = 49, NB = 24, H = 256, Tout = 25, NG = 5;
constexpr int KCH = 8;                        // H/32 K-chunks per WMMA sweep
constexpr int WAVES = 16;
constexpr int THREADS = WAVES * 32;           // 512
constexpr int PERBONE = NG * H * H;           // 327,680 elems (one bone, one matrix)
constexpr long PERMAT = (long)NB * PERBONE;   // 7,864,320 elems
constexpr int BH = B * H;                     // 4096
constexpr int STATE = NB * BH;                // 98,304
constexpr int OUTSTRIDE_B = Tout * NB * H;    // per-batch stride of output tensor
constexpr long OUTHALF = (long)B * OUTSTRIDE_B; // 2,457,600
}

DEV_INLINE unsigned short f2bf(float f) {
  unsigned u = __float_as_uint(f);
  u += 0x7FFFu + ((u >> 16) & 1u);   // round-to-nearest-even
  return (unsigned short)(u >> 16);
}
DEV_INLINE float sigm(float x)  { return 1.0f / (1.0f + __expf(-x)); }
DEV_INLINE float tanh_(float x) { return 1.0f - 2.0f / (__expf(2.0f * x) + 1.0f); }

// A-fragment: 16-bit A 16x32 layout (ISA 7.12.2): lane%16 = M row, lane/16 picks
// K half; slots 0-7 = 8 contiguous K at (kc*32 + khalf*8), slots 8-15 at +16.
DEV_INLINE v16bf load_a_frag(const unsigned short* p) {
  v8bf lo, hi;
  __builtin_memcpy(&lo, p, 16);
  __builtin_memcpy(&hi, p + 16, 16);
  v16bf r;
  #pragma unroll
  for (int i = 0; i < 8; ++i) { r[i] = lo[i]; r[i + 8] = hi[i]; }
  return r;
}
// B-fragment: pre-swizzled in workspace so each lane reads 16 contiguous bf16.
DEV_INLINE v16bf load_b_frag(const unsigned short* __restrict__ p) {
  v16bf r;
  __builtin_memcpy(&r, p, 32);
  return r;
}

// ---------------------------------------------------------------------------
// Kernel 1: convert f32 weights -> bf16, swizzled into WMMA B-fragment order.
// dst layout per matrix: [bone][nt=80][kc=8][lane=32][slot=16]
// B 32x16 striping: lane%16 = N column, lane/16 = K half; slot s -> k_local =
// (lane/16)*8 + (s&7) + (s>>3)*16  (matches A-fragment order).
// ---------------------------------------------------------------------------
__global__ void swizzle_weights(const float* __restrict__ U1, const float* __restrict__ Wt1,
                                const float* __restrict__ Ws1, const float* __restrict__ U2,
                                const float* __restrict__ Wt2, const float* __restrict__ Ws2,
                                unsigned short* __restrict__ dst) {
  using namespace cfg;
  const long total = 6 * PERMAT;
  for (long idx = (long)blockIdx.x * blockDim.x + threadIdx.x; idx < total;
       idx += (long)gridDim.x * blockDim.x) {
    const int m  = (int)(idx / PERMAT);
    const int r  = (int)(idx - (long)m * PERMAT);
    const int nb = r / PERBONE;
    const int q  = r - nb * PERBONE;
    const int s  = q & 15;
    const int l  = (q >> 4) & 31;
    const int kc = (q >> 9) & 7;
    const int nt = q >> 12;                 // 0..79
    const int k    = kc * 32 + (l >> 4) * 8 + (s & 7) + (s >> 3) * 16;
    const int ncol = nt * 16 + (l & 15);
    const int g = ncol >> 8, d = ncol & 255;
    const float* src = (m == 0) ? U1 : (m == 1) ? Wt1 : (m == 2) ? Ws1
                     : (m == 3) ? U2 : (m == 4) ? Wt2 : Ws2;
    dst[idx] = f2bf(src[(((long)nb * NG + g) * H + k) * H + d]);
  }
}

// ---------------------------------------------------------------------------
// Kernel 2: per-(b,h) boundary means: cs_mean, hs1_init, hs2_init
// ---------------------------------------------------------------------------
__global__ void init_bounds(const float* __restrict__ hidden, const float* __restrict__ cells,
                            const float* __restrict__ gss,
                            float* __restrict__ cs_mean, float* __restrict__ hs1i,
                            float* __restrict__ hs2i) {
  using namespace cfg;
  const int id = blockIdx.x * blockDim.x + threadIdx.x;   // 0..4095
  if (id >= BH) return;
  const int b = id >> 8, h = id & 255;
  float hsum = 0.f, csum = 0.f, gsum = 0.f;
  for (int t = 0; t < Tin; ++t) {
    gsum += gss[((long)b * Tin + t) * H + h];
    for (int n = 0; n < NB; ++n) {
      const long idx = (((long)b * Tin + t) * NB + n) * H + h;
      hsum += hidden[idx];
      csum += cells[idx];
    }
  }
  cs_mean[id] = csum * (1.0f / (Tin * NB));
  hs1i[id]    = hsum * (1.0f / (Tin * NB));
  hs2i[id]    = (hsum + gsum) * (1.0f / (Tin * (NB + 1)));
}

// ---------------------------------------------------------------------------
// Kernel 3: per-(n,b,h) row-0 states h1,c1,h2,c2  (state layout [NB][B][H])
// ---------------------------------------------------------------------------
__global__ void init_states(const float* __restrict__ hidden, const float* __restrict__ cells,
                            const float* __restrict__ gts,
                            float* __restrict__ h1, float* __restrict__ c1,
                            float* __restrict__ h2, float* __restrict__ c2) {
  using namespace cfg;
  const int id = blockIdx.x * blockDim.x + threadIdx.x;   // 0..98303
  if (id >= STATE) return;
  const int h = id & 255, b = (id >> 8) & 15, n = id >> 12;
  float hsum = 0.f, csum = 0.f;
  for (int t = 0; t < Tin; ++t) {
    const long idx = (((long)b * Tin + t) * NB + n) * H + h;
    hsum += hidden[idx];
    csum += cells[idx];
  }
  const float c0 = csum * (1.0f / Tin);
  h1[id] = hsum * (1.0f / Tin);
  c1[id] = c0;
  c2[id] = c0;
  h2[id] = (gts[((long)b * NB + n) * H + h] + hsum) * (1.0f / (Tin + 1));
}

// ---------------------------------------------------------------------------
// One ST-LSTM cell: g = x*U + h_t*Wt + h_s*Ws + b, gates, state update.
// M=16 (batch), K=256, N=1280. Wave w owns hidden columns [16w,16w+16)
// across all 5 gates => epilogue (gates + next-operand staging) is wave-local.
// ---------------------------------------------------------------------------
DEV_INLINE void cell_step(
    int lane, int w,
    const unsigned short* __restrict__ Bu, const unsigned short* __restrict__ Bwt,
    const unsigned short* __restrict__ Bws, const float* __restrict__ bias,
    const unsigned short* xA, const unsigned short* htA, const unsigned short* hsA,
    float* csL,
    const float* __restrict__ ctG, float* __restrict__ cG, float* __restrict__ hG,
    unsigned short* hsW, unsigned short* xW,
    unsigned short* htStage, const float* __restrict__ htStageSrc,
    unsigned short* xStage, const float* __restrict__ xStageSrc, long xStageStride,
    float* __restrict__ outH, float* __restrict__ outC) {
  using namespace cfg;
  const int nlocal = lane & 15;
  const int khalf  = lane >> 4;
  const v8f zero = {0.f, 0.f, 0.f, 0.f, 0.f, 0.f, 0.f, 0.f};
  v8f acc[NG];
  #pragma unroll
  for (int g = 0; g < NG; ++g) acc[g] = zero;

  #pragma unroll
  for (int kc = 0; kc < KCH; ++kc) {
    const int arow = nlocal * H + kc * 32 + khalf * 8;
    v16bf aX = load_a_frag(xA  + arow);
    v16bf aT = load_a_frag(htA + arow);
    v16bf aS = load_a_frag(hsA + arow);
    #pragma unroll
    for (int g = 0; g < NG; ++g) {
      const int ntg = g * 16 + w;
      const unsigned foff = ((unsigned)(ntg * KCH + kc) * 32 + (unsigned)lane) * 16u;
      v16bf bU = load_b_frag(Bu  + foff);
      v16bf bT = load_b_frag(Bwt + foff);
      v16bf bS = load_b_frag(Bws + foff);
      acc[g] = __builtin_amdgcn_wmma_f32_16x16x32_bf16(false, aX, false, bU, (short)0, acc[g], false, false);
      acc[g] = __builtin_amdgcn_wmma_f32_16x16x32_bf16(false, aT, false, bT, (short)0, acc[g], false, false);
      acc[g] = __builtin_amdgcn_wmma_f32_16x16x32_bf16(false, aS, false, bS, (short)0, acc[g], false, false);
    }
  }
  __syncthreads();   // all waves done reading shared A operands before epilogue writes

  const int hc = w * 16 + nlocal;
  #pragma unroll
  for (int g = 0; g < NG; ++g) {
    const float bv = bias[g * H + hc];
    #pragma unroll
    for (int v = 0; v < 8; ++v) acc[g][v] += bv;
  }
  // D layout: VGPR v, lanes 0-15 -> M=v, lanes 16-31 -> M=v+8; N = lane%16.
  const int mbase = khalf * 8;
  #pragma unroll
  for (int v = 0; v < 8; ++v) {
    const int m   = mbase + v;
    const int off = m * H + hc;
    const float ct = ctG[off];
    const float cs = csL[off];
    const float iv = sigm(acc[0][v]);
    const float fs = sigm(acc[1][v]);
    const float ft = sigm(acc[2][v]);
    const float ov = sigm(acc[3][v]);
    const float cn = tanh_(acc[4][v]);
    const float ch = iv * cn + ft * ct + fs * cs;
    const float hv = ov * tanh_(ch);
    csL[off] = ch;        // running spatial c for next bone (wave-local)
    cG[off]  = ch;        // temporal c state for next frame
    hG[off]  = hv;        // temporal h state for next frame
    const unsigned short hb = f2bf(hv);
    hsW[off] = hb;        // running spatial h (A operand, LDS bf16)
    if (xW) xW[off] = hb; // layer-1 h feeds layer-2 x
    // fused staging of the NEXT cell's LDS operands (critical-path trim)
    if (htStageSrc) htStage[off] = f2bf(htStageSrc[off]);
    if (xStageSrc)  xStage[off]  = f2bf(xStageSrc[(long)m * xStageStride + hc]);
    if (outH) {
      __builtin_nontemporal_store(hv, &outH[(long)m * OUTSTRIDE_B + hc]);
      __builtin_nontemporal_store(ch, &outC[(long)m * OUTSTRIDE_B + hc]);
    }
  }
}

// ---------------------------------------------------------------------------
// Kernel 4: persistent recurrence — 1 workgroup, 16 waves, 25 frames x 24
// bones x 2 layers of serially-dependent cells. Operand staging for cell k+1
// is folded into cell k's epilogue, so the only per-cell overhead between
// WMMA bursts is two workgroup barriers.
// ---------------------------------------------------------------------------
__global__ __launch_bounds__(cfg::THREADS, 1)
void st_lstm_recurrence(const unsigned short* __restrict__ wswz,
                        const float* __restrict__ b1, const float* __restrict__ b2,
                        const float* __restrict__ p,
                        const float* __restrict__ cs_mean,
                        const float* __restrict__ hs1i, const float* __restrict__ hs2i,
                        float* __restrict__ h1, float* __restrict__ c1,
                        float* __restrict__ h2, float* __restrict__ c2,
                        float* __restrict__ outH, float* __restrict__ outC) {
  using namespace cfg;
  __shared__ unsigned short xA[BH];    // x operand, bf16 [16][256]
  __shared__ unsigned short htA[BH];   // temporal h operand
  __shared__ unsigned short hs1A[BH];  // running spatial h, layer 1
  __shared__ unsigned short hs2A[BH];  // running spatial h, layer 2
  __shared__ float c1s[BH];            // running spatial c, layer 1
  __shared__ float c2s[BH];            // running spatial c, layer 2

  const int tid  = threadIdx.x;
  const int lane = tid & 31;
  const int w    = tid >> 5;

  // initial staging for (t=0, n=0): x = p[:,0,0,:], h_t = h1[0]
  for (int i = tid; i < BH; i += THREADS) {
    const int b = i >> 8, h = i & 255;
    xA[i]  = f2bf(p[(long)b * Tin * NB * H + h]);
    htA[i] = f2bf(h1[i]);
  }

  for (int t = 0; t < Tout; ++t) {
    __syncthreads();   // cover initial staging / previous frame's epilogues
    // per-frame spatial boundary reset
    for (int i = tid; i < BH; i += THREADS) {
      hs1A[i] = f2bf(hs1i[i]);
      hs2A[i] = f2bf(hs2i[i]);
      const float cm = cs_mean[i];
      c1s[i] = cm;
      c2s[i] = cm;
    }

    for (int n = 0; n < NB; ++n) {
      const long bo = (long)n * PERBONE;
      const int  so = n * BH;
      __syncthreads();   // staging + frame-init visible before operand reads

      // layer 1; epilogue also stages layer-2's h_t (= old h2[n]) into htA
      cell_step(lane, w,
                wswz + 0 * PERMAT + bo, wswz + 1 * PERMAT + bo, wswz + 2 * PERMAT + bo,
                b1 + n * NG * H,
                xA, htA, hs1A, c1s,
                c1 + so, c1 + so, h1 + so,
                hs1A, xA,
                htA, h2 + so,
                nullptr, nullptr, 0,
                nullptr, nullptr);
      __syncthreads();

      // next-cell staging sources (next bone, or bone 0 of next frame)
      const float* nxHt = nullptr;
      const float* nxX  = nullptr;
      long nxStride = H;
      if (n + 1 < NB) {
        nxHt = h1 + (n + 1) * BH;
        if (t == 0) { nxX = p + (long)(n + 1) * H; nxStride = (long)Tin * NB * H; }
        else        { nxX = h2 + (n + 1) * BH; }
      } else if (t + 1 < Tout) {
        nxHt = h1;          // bone 0 state, rewritten earlier this frame
        nxX  = h2;          // next frame has t>=1 so x comes from h2
      }

      // layer 2; epilogue stages next cell's x / h_t and streams outputs
      cell_step(lane, w,
                wswz + 3 * PERMAT + bo, wswz + 4 * PERMAT + bo, wswz + 5 * PERMAT + bo,
                b2 + n * NG * H,
                xA, htA, hs2A, c2s,
                c2 + so, c2 + so, h2 + so,
                hs2A, nullptr,
                htA, nxHt,
                xA, nxX, nxStride,
                outH + ((long)t * NB + n) * H, outC + ((long)t * NB + n) * H);
    }
  }
}

// ---------------------------------------------------------------------------
extern "C" void kernel_launch(void* const* d_in, const int* in_sizes, int n_in,
                              void* d_out, int out_size, void* d_ws, size_t ws_size,
                              hipStream_t stream) {
  using namespace cfg;
  const float* hidden = (const float*)d_in[0];
  const float* cells  = (const float*)d_in[1];
  const float* gts    = (const float*)d_in[2];
  const float* gss    = (const float*)d_in[3];
  const float* p      = (const float*)d_in[4];
  const float* U1  = (const float*)d_in[5];
  const float* Wt1 = (const float*)d_in[6];
  const float* Ws1 = (const float*)d_in[7];
  const float* b1  = (const float*)d_in[8];
  const float* U2  = (const float*)d_in[9];
  const float* Wt2 = (const float*)d_in[10];
  const float* Ws2 = (const float*)d_in[11];
  const float* b2  = (const float*)d_in[12];
  float* out = (float*)d_out;

  // workspace layout: [6 bf16 swizzled matrices | boundary f32 | 4 state f32]
  unsigned short* wswz = (unsigned short*)d_ws;
  float* fws = (float*)((char*)d_ws + 6ull * (size_t)PERMAT * sizeof(unsigned short));
  float* cs_mean = fws;
  float* hs1i    = fws + BH;
  float* hs2i    = fws + 2 * BH;
  float* h1 = fws + 3 * BH;
  float* c1 = h1 + STATE;
  float* h2 = c1 + STATE;
  float* c2 = h2 + STATE;

  swizzle_weights<<<2048, 256, 0, stream>>>(U1, Wt1, Ws1, U2, Wt2, Ws2, wswz);
  init_bounds<<<(BH + 255) / 256, 256, 0, stream>>>(hidden, cells, gss, cs_mean, hs1i, hs2i);
  init_states<<<(STATE + 255) / 256, 256, 0, stream>>>(hidden, cells, gts, h1, c1, h2, c2);
  st_lstm_recurrence<<<1, THREADS, 0, stream>>>(wswz, b1, b2, p, cs_mean, hs1i, hs2i,
                                                h1, c1, h2, c2, out, out + OUTHALF);
}
